// MambaLM_21672404976175
// MI455X (gfx1250) — compile-verified, matrix operand order checked by reading, not run on hardware
//
#include <hip/hip_runtime.h>
#include <hip/hip_bf16.h>

// ---------------------------------------------------------------------------
// Mamba LM forward (2 layers) for gfx1250.
// Big GEMMs (in_proj, out_proj, lm_head): v_wmma_f32_16x16x32_bf16 with
// double-buffered LDS tiles fed by global_load_async_to_lds_b128 (ASYNCcnt).
// Wave tile 64x64 -> 16 ds_load_b128 : 16 v_wmma per K-step.
// Selective scan: lane-per-(channel,state) with __shfl_xor 16-lane reduction.
// Everything else: fp32 VALU.
// ---------------------------------------------------------------------------

#define VOCAB   32000
#define D_MODEL 1024
#define N_LAYERS 2
#define D_STATE 16
#define D_CONV  4
#define D_INNER 2048
#define DT_RANK 64
#define SEQ     2048

typedef __attribute__((ext_vector_type(16))) __bf16 v16bf;
typedef __attribute__((ext_vector_type(8)))  __bf16 v8bf;
typedef __attribute__((ext_vector_type(8)))  float  v8f;

// ---------------------------------------------------------------------------
// Embedding gather: x[l, d] = emb[ids[l], d]
// ---------------------------------------------------------------------------
__global__ void embed_kernel(const int* __restrict__ ids,
                             const float* __restrict__ emb,
                             float* __restrict__ x) {
    int i = blockIdx.x * 256 + threadIdx.x;           // i in [0, SEQ*D_MODEL)
    int l = i >> 10;                                  // / D_MODEL
    int d = i & 1023;
    x[i] = emb[(size_t)ids[l] * D_MODEL + d];
}

// ---------------------------------------------------------------------------
// LayerNorm over rows of length D_MODEL, output bf16.
// grid = SEQ blocks, block = 256 threads.
// ---------------------------------------------------------------------------
__global__ void layernorm_bf16_kernel(const float* __restrict__ x,
                                      const float* __restrict__ g,
                                      const float* __restrict__ b,
                                      unsigned short* __restrict__ outu) {
    __shared__ float s1[256];
    __shared__ float s2[256];
    __bf16* out = (__bf16*)outu;
    int row = blockIdx.x;
    int tid = threadIdx.x;
    const float* p = x + (size_t)row * D_MODEL;
    float sum = 0.f, sq = 0.f;
    for (int i = tid; i < D_MODEL; i += 256) {
        float v = p[i];
        sum += v; sq += v * v;
    }
    s1[tid] = sum; s2[tid] = sq;
    __syncthreads();
    for (int off = 128; off > 0; off >>= 1) {
        if (tid < off) { s1[tid] += s1[tid + off]; s2[tid] += s2[tid + off]; }
        __syncthreads();
    }
    float mean = s1[0] * (1.0f / D_MODEL);
    float var  = s2[0] * (1.0f / D_MODEL) - mean * mean;
    float rs   = rsqrtf(var + 1e-5f);
    for (int i = tid; i < D_MODEL; i += 256) {
        float v = (p[i] - mean) * rs * g[i] + b[i];
        out[(size_t)row * D_MODEL + i] = (__bf16)v;
    }
}

// ---------------------------------------------------------------------------
// fp32 -> bf16 conversion (weights / activations)
// ---------------------------------------------------------------------------
__global__ void f32_to_bf16_kernel(const float* __restrict__ in,
                                   unsigned short* __restrict__ outu, int n) {
    int i = blockIdx.x * 256 + threadIdx.x;
    if (i < n) ((__bf16*)outu)[i] = (__bf16)in[i];
}

// ---------------------------------------------------------------------------
// bf16 WMMA GEMM:  C[m,n] = sum_k A[m,k] * B[n,k]  (+ add[m,n] if HAS_ADD)
// A: M x K bf16 row-major.  B: N x K bf16 row-major (A * B^T).
// Grid: (N/256, M/128), block 256 threads = 8 waves (2 M x 4 N),
// wave tile 64x64 = 4x4 WMMA 16x16 tiles.
//
// Per 32-wide K step, the 128x32 A and 256x32 B tiles are staged in LDS via
// global_load_async_to_lds_b128 (ASYNCcnt), double-buffered so the async
// copy of step k+1 overlaps the ds_load_b128 + v_wmma of step k.
//
// Fragment layout (CDNA5 ISA, 16-bit A/B): lane r = lane&15 owns row r of
// the 16-row tile; lane-half h owns K in [8h,8h+8) U [16+8h,16+8h+8): two
// contiguous 16-byte reads per fragment (identical addressing in LDS tiles).
// ---------------------------------------------------------------------------
template <bool HAS_ADD>
__global__ __launch_bounds__(256)
void gemm_bf16_wmma_kernel(const unsigned short* __restrict__ Au,
                           const unsigned short* __restrict__ Bu,
                           const float* __restrict__ add,
                           float* __restrict__ C,
                           int M, int N, int K) {
    const __bf16* A = (const __bf16*)Au;
    const __bf16* B = (const __bf16*)Bu;

    // Double-buffered tiles: A 128x32 (8KB), B 256x32 (16KB) -> 48KB LDS.
    __shared__ __align__(16) __bf16 As[2][128 * 32];
    __shared__ __align__(16) __bf16 Bs[2][256 * 32];

    const int tid  = threadIdx.x;
    const int lane = tid & 31;
    const int wave = tid >> 5;               // 0..7
    const int lr   = lane & 15;
    const int hl   = lane >> 4;              // lane half: 0 or 1

    const int bm = blockIdx.y * 128;
    const int bn = blockIdx.x * 256;
    const int wmLoc = (wave >> 2) * 64;      // wave M origin within block tile
    const int wnLoc = (wave & 3) * 64;       // wave N origin within block tile

    // Issue the async global->LDS copy of one 32-wide K slice of both tiles.
    // A tile: 128 rows x 64B = 512 16B-chunks (2/thread).
    // B tile: 256 rows x 64B = 1024 16B-chunks (4/thread).
    auto issue_tiles = [&](int buf, int k) {
        unsigned aB = (unsigned)(uintptr_t)&As[buf][0];
        unsigned bB = (unsigned)(uintptr_t)&Bs[buf][0];
#pragma unroll
        for (int c = 0; c < 512; c += 256) {
            int ch  = c + tid;
            int row = ch >> 2;
            int cc  = ch & 3;
            const __bf16* ga = A + (size_t)(bm + row) * K + k + cc * 8;
            unsigned la = aB + (unsigned)(row * 64 + cc * 16);
            asm volatile("global_load_async_to_lds_b128 %0, %1, off"
                         :: "v"(la), "v"(ga) : "memory");
        }
#pragma unroll
        for (int c = 0; c < 1024; c += 256) {
            int ch  = c + tid;
            int row = ch >> 2;
            int cc  = ch & 3;
            const __bf16* gb = B + (size_t)(bn + row) * K + k + cc * 8;
            unsigned lb = bB + (unsigned)(row * 64 + cc * 16);
            asm volatile("global_load_async_to_lds_b128 %0, %1, off"
                         :: "v"(lb), "v"(gb) : "memory");
        }
    };

    v8f acc[4][4];
#pragma unroll
    for (int mt = 0; mt < 4; ++mt)
#pragma unroll
        for (int nt = 0; nt < 4; ++nt)
            acc[mt][nt] = (v8f)(0.0f);

    const int KT = K >> 5;                   // K / 32
    issue_tiles(0, 0);
    asm volatile("s_wait_asynccnt 0" ::: "memory");
    __syncthreads();

    for (int kt = 0; kt < KT; ++kt) {
        const int buf = kt & 1;
        if (kt + 1 < KT) issue_tiles(buf ^ 1, (kt + 1) * 32);

        union Frag { v16bf v; v8bf h[2]; };
        Frag a[4], b[4];
#pragma unroll
        for (int mt = 0; mt < 4; ++mt) {
            const __bf16* ap = &As[buf][(wmLoc + mt * 16 + lr) * 32 + hl * 8];
            a[mt].h[0] = *(const v8bf*)ap;
            a[mt].h[1] = *(const v8bf*)(ap + 16);
        }
#pragma unroll
        for (int nt = 0; nt < 4; ++nt) {
            const __bf16* bp = &Bs[buf][(wnLoc + nt * 16 + lr) * 32 + hl * 8];
            b[nt].h[0] = *(const v8bf*)bp;
            b[nt].h[1] = *(const v8bf*)(bp + 16);
        }
#pragma unroll
        for (int mt = 0; mt < 4; ++mt)
#pragma unroll
            for (int nt = 0; nt < 4; ++nt)
                acc[mt][nt] = __builtin_amdgcn_wmma_f32_16x16x32_bf16(
                    /*neg_a=*/false, a[mt].v,
                    /*neg_b=*/false, b[nt].v,
                    /*c_mod=*/(short)0, acc[mt][nt],
                    /*reuse_a=*/false, /*reuse_b=*/false);

        if (kt + 1 < KT)
            asm volatile("s_wait_asynccnt 0" ::: "memory");
        __syncthreads();
    }

    // Store D: VGPR i -> (M = i + 8*hl, N = lr) within each 16x16 tile.
#pragma unroll
    for (int mt = 0; mt < 4; ++mt) {
#pragma unroll
        for (int nt = 0; nt < 4; ++nt) {
#pragma unroll
            for (int i = 0; i < 8; ++i) {
                int row = bm + wmLoc + mt * 16 + i + 8 * hl;
                int col = bn + wnLoc + nt * 16 + lr;
                size_t idx = (size_t)row * N + col;
                float v = acc[mt][nt][i];
                if (HAS_ADD) v += add[idx];
                C[idx] = v;
            }
        }
    }
}

// ---------------------------------------------------------------------------
// Depthwise causal conv (D_CONV=4) + SiLU.
// xi[l,d] = xz[l*2*D_INNER + d]; out xc[l,d].
// ---------------------------------------------------------------------------
__global__ void conv_silu_kernel(const float* __restrict__ xz,
                                 const float* __restrict__ cw,
                                 const float* __restrict__ cb,
                                 float* __restrict__ xc) {
    int i = blockIdx.x * 256 + threadIdx.x;    // [0, SEQ*D_INNER)
    int l = i >> 11;                            // / D_INNER
    int d = i & (D_INNER - 1);
    float s = cb[d];
#pragma unroll
    for (int k = 0; k < D_CONV; ++k) {
        int li = l - (D_CONV - 1) + k;
        if (li >= 0)
            s += xz[(size_t)li * (2 * D_INNER) + d] * cw[d * D_CONV + k];
    }
    float r = s / (1.0f + __expf(-s));          // silu
    xc[(size_t)l * D_INNER + d] = r;
}

// ---------------------------------------------------------------------------
// x_proj: xdbl[l, r] = sum_d xc[l,d] * xpw[r,d]   (r < 96)
// grid = SEQ blocks, block = 128 (threads 96..127 only help stage LDS).
// ---------------------------------------------------------------------------
__global__ void xproj_kernel(const float* __restrict__ xc,
                             const float* __restrict__ xpw,
                             float* __restrict__ xdbl) {
    __shared__ float s[D_INNER];
    int l = blockIdx.x;
    const float* row = xc + (size_t)l * D_INNER;
    for (int i = threadIdx.x; i < D_INNER; i += 128) s[i] = row[i];
    __syncthreads();
    int r = threadIdx.x;
    if (r < (DT_RANK + 2 * D_STATE)) {
        const float* wr = xpw + (size_t)r * D_INNER;
        float acc = 0.f;
        for (int d = 0; d < D_INNER; ++d) acc += s[d] * wr[d];
        xdbl[(size_t)l * 96 + r] = acc;
    }
}

// ---------------------------------------------------------------------------
// dt_proj + softplus: dt[l,d] = softplus(sum_r xdbl[l,r]*dtw[d,r] + dtb[d])
// grid = (D_INNER/256, SEQ), block = 256.
// ---------------------------------------------------------------------------
__global__ void dtproj_kernel(const float* __restrict__ xdbl,
                              const float* __restrict__ dtw,
                              const float* __restrict__ dtb,
                              float* __restrict__ dt) {
    __shared__ float s[DT_RANK];
    int l = blockIdx.y;
    if (threadIdx.x < DT_RANK) s[threadIdx.x] = xdbl[(size_t)l * 96 + threadIdx.x];
    __syncthreads();
    int d = blockIdx.x * 256 + threadIdx.x;
    const float* wr = dtw + (size_t)d * DT_RANK;
    float acc = dtb[d];
#pragma unroll 8
    for (int r = 0; r < DT_RANK; ++r) acc += s[r] * wr[r];
    float sp = (acc > 20.f) ? acc : log1pf(__expf(acc));
    dt[(size_t)l * D_INNER + d] = sp;
}

// ---------------------------------------------------------------------------
// Selective scan, lane-per-(channel, state).
// Global lane g -> channel d = g>>4, state n = g&15.  Each lane carries one
// recurrent state h; per step: one exp + fma, then the 16-state reduction
// y = sum_n h*C is 4 __shfl_xor steps (stays inside the 16-lane group of a
// wave32).  32768 lanes = 1024 waves.
// grid = D_INNER*16/256 = 128 blocks, block = 256.
// ---------------------------------------------------------------------------
__global__ void scan_kernel(const float* __restrict__ dt,
                            const float* __restrict__ xc,
                            const float* __restrict__ xdbl,
                            const float* __restrict__ alog,
                            float* __restrict__ ys) {
    int g = blockIdx.x * 256 + threadIdx.x;   // [0, D_INNER*D_STATE)
    int d = g >> 4;
    int n = g & 15;
    float A = -__expf(alog[(size_t)d * D_STATE + n]);
    float h = 0.f;
    for (int l = 0; l < SEQ; ++l) {
        float dtv = dt[(size_t)l * D_INNER + d];
        float xcv = xc[(size_t)l * D_INNER + d];
        float Bv  = xdbl[(size_t)l * 96 + DT_RANK + n];
        float Cv  = xdbl[(size_t)l * 96 + DT_RANK + D_STATE + n];
        h = __expf(dtv * A) * h + (dtv * Bv) * xcv;
        float p = h * Cv;
        p += __shfl_xor(p, 1);
        p += __shfl_xor(p, 2);
        p += __shfl_xor(p, 4);
        p += __shfl_xor(p, 8);
        if (n == 0) ys[(size_t)l * D_INNER + d] = p;
    }
}

// ---------------------------------------------------------------------------
// Gate: y = (ys + D_skip*xc) * silu(z); output bf16 for the out_proj GEMM.
// z[l,d] = xz[l*2*D_INNER + D_INNER + d]
// ---------------------------------------------------------------------------
__global__ void gate_kernel(const float* __restrict__ ys,
                            const float* __restrict__ xc,
                            const float* __restrict__ xz,
                            const float* __restrict__ dsk,
                            unsigned short* __restrict__ ybfu) {
    int i = blockIdx.x * 256 + threadIdx.x;    // [0, SEQ*D_INNER)
    int l = i >> 11;
    int d = i & (D_INNER - 1);
    float z = xz[(size_t)l * (2 * D_INNER) + D_INNER + d];
    float sz = z / (1.0f + __expf(-z));
    float y = (ys[i] + dsk[d] * xc[i]) * sz;
    ((__bf16*)ybfu)[i] = (__bf16)y;
}

// ---------------------------------------------------------------------------
// Host-side launch
// ---------------------------------------------------------------------------
extern "C" void kernel_launch(void* const* d_in, const int* in_sizes, int n_in,
                              void* d_out, int out_size, void* d_ws, size_t ws_size,
                              hipStream_t stream) {
    (void)in_sizes; (void)n_in; (void)out_size; (void)ws_size;

    const int*   ids   = (const int*)  d_in[0];
    const float* emb   = (const float*)d_in[1];
    const float* ln_g  = (const float*)d_in[2];
    const float* ln_b  = (const float*)d_in[3];
    const float* inw   = (const float*)d_in[4];
    const float* cw    = (const float*)d_in[5];
    const float* cb    = (const float*)d_in[6];
    const float* xpw   = (const float*)d_in[7];
    const float* dtw   = (const float*)d_in[8];
    const float* dtb   = (const float*)d_in[9];
    const float* alog  = (const float*)d_in[10];
    const float* dsk   = (const float*)d_in[11];
    const float* outw  = (const float*)d_in[12];
    const float* lnfg  = (const float*)d_in[13];
    const float* lnfb  = (const float*)d_in[14];
    const float* headw = (const float*)d_in[15];
    float* logits = (float*)d_out;

    // Workspace carve (256B aligned), ~166 MB total.
    size_t off = 0;
    char* base = (char*)d_ws;
    auto alloc = [&](size_t bytes) -> void* {
        void* p = base + off;
        off += (bytes + 255) & ~(size_t)255;
        return p;
    };
    float*          x    = (float*)         alloc((size_t)SEQ * D_MODEL * 4);
    unsigned short* hb   = (unsigned short*)alloc((size_t)SEQ * D_MODEL * 2);
    unsigned short* wbf  = (unsigned short*)alloc((size_t)VOCAB * D_MODEL * 2);
    float*          xz   = (float*)         alloc((size_t)SEQ * 2 * D_INNER * 4);
    float*          xc   = (float*)         alloc((size_t)SEQ * D_INNER * 4);
    float*          xdbl = (float*)         alloc((size_t)SEQ * 96 * 4);
    float*          dt   = (float*)         alloc((size_t)SEQ * D_INNER * 4);
    float*          ys   = (float*)         alloc((size_t)SEQ * D_INNER * 4);
    unsigned short* ybf  = (unsigned short*)alloc((size_t)SEQ * D_INNER * 2);

    // 1) Embedding
    embed_kernel<<<(SEQ * D_MODEL) / 256, 256, 0, stream>>>(ids, emb, x);

    for (int l = 0; l < N_LAYERS; ++l) {
        // 2) LN -> bf16
        layernorm_bf16_kernel<<<SEQ, 256, 0, stream>>>(
            x, ln_g + (size_t)l * D_MODEL, ln_b + (size_t)l * D_MODEL, hb);

        // 3) in_proj: xz[2048 x 4096] = hb * inw^T (WMMA bf16)
        {
            int n = 2 * D_INNER * D_MODEL;
            f32_to_bf16_kernel<<<(n + 255) / 256, 256, 0, stream>>>(
                inw + (size_t)l * n, wbf, n);
            gemm_bf16_wmma_kernel<false>
                <<<dim3((2 * D_INNER) / 256, SEQ / 128), 256, 0, stream>>>(
                hb, wbf, nullptr, xz, SEQ, 2 * D_INNER, D_MODEL);
        }

        // 4) causal conv + silu
        conv_silu_kernel<<<(SEQ * D_INNER) / 256, 256, 0, stream>>>(
            xz, cw + (size_t)l * D_INNER * D_CONV, cb + (size_t)l * D_INNER, xc);

        // 5) x_proj (N=96, fp32)
        xproj_kernel<<<SEQ, 128, 0, stream>>>(
            xc, xpw + (size_t)l * 96 * D_INNER, xdbl);

        // 6) dt_proj + softplus
        dtproj_kernel<<<dim3(D_INNER / 256, SEQ), 256, 0, stream>>>(
            xdbl, dtw + (size_t)l * D_INNER * DT_RANK, dtb + (size_t)l * D_INNER, dt);

        // 7) selective scan (lane per (d, n))
        scan_kernel<<<(D_INNER * D_STATE) / 256, 256, 0, stream>>>(
            dt, xc, xdbl, alog + (size_t)l * D_INNER * D_STATE, ys);

        // 8) gate -> bf16
        gate_kernel<<<(SEQ * D_INNER) / 256, 256, 0, stream>>>(
            ys, xc, xz, dsk + (size_t)l * D_INNER, ybf);

        // 9) out_proj + residual: x = x + ybf * outw^T (WMMA bf16)
        {
            int n = D_MODEL * D_INNER;
            f32_to_bf16_kernel<<<(n + 255) / 256, 256, 0, stream>>>(
                outw + (size_t)l * n, wbf, n);
            gemm_bf16_wmma_kernel<true>
                <<<dim3(D_MODEL / 256, SEQ / 128), 256, 0, stream>>>(
                ybf, wbf, x, x, SEQ, D_MODEL, D_INNER);
        }
    }

    // 10) final LN -> bf16
    layernorm_bf16_kernel<<<SEQ, 256, 0, stream>>>(x, lnfg, lnfb, hb);

    // 11) LM head: logits[2048 x 32000] = hb * headw^T (WMMA bf16)
    {
        int n = VOCAB * D_MODEL;
        f32_to_bf16_kernel<<<(n + 255) / 256, 256, 0, stream>>>(headw, wbf, n);
        gemm_bf16_wmma_kernel<false>
            <<<dim3(VOCAB / 256, SEQ / 128), 256, 0, stream>>>(
            hb, wbf, nullptr, logits, SEQ, VOCAB, D_MODEL);
    }
}
